// TreeNetwork_33913061769574
// MI455X (gfx1250) — matrix operand order repaired and checked by reading, not compiled, not run on hardware
//
#include <hip/hip_runtime.h>
#include <hip/hip_bf16.h>
#include <math.h>

// ---------------------------------------------------------------------------
// TreeNetwork on MI455X (gfx1250, wave32, WMMA + TDM)
//
// reference:  h = tanh(emb[leaf_ids])                  [64,2048,512]
//             11x: h = tanh(view(h,[N/2,1024]) @ W^T + b)
//
// concat(left,right) of adjacent siblings is a reinterpret of the row-major
// [N,512] buffer as [N/2,1024] -> every level is a dense GEMM. Intermediates
// in bf16 (tanh output in [-1,1]); v_wmma_f32_16x16x32_bf16 with f32 accum.
// W tiles are staged into LDS by the Tensor Data Mover (tensor_load_to_lds),
// double-buffered and overlapped with WMMA compute; TENSORcnt tracks the DMA.
// An empty asm with a "memory" clobber + escaped LDS pointer tells the
// optimizer that the TDM wrote the LDS tile (the builtin's descriptor carries
// the LDS address as plain integers, so LLVM cannot see the aliasing itself).
// ---------------------------------------------------------------------------

#define HID        512
#define KDIM       1024           // 2*HID
#define B_TREES    64
#define LEAVES     2048
#define NLEAF      (B_TREES * LEAVES)   // 131072 leaf nodes
#define KC         256            // K-chunk of W staged in LDS per iteration
#define LDS_STRIDE (KC + 8)       // 528B padded LDS row (TDM pad fields match)
#define WTILE_H    (64 * LDS_STRIDE)    // halves per LDS W-tile buffer

typedef __bf16 bf16_t;
typedef __attribute__((ext_vector_type(16))) bf16_t v16bf;
typedef __attribute__((ext_vector_type(8)))  bf16_t v8bf;
typedef __attribute__((ext_vector_type(8)))  float  v8f;
typedef __attribute__((ext_vector_type(4)))  unsigned int u32x4;
typedef __attribute__((ext_vector_type(8)))  int i32x8;
typedef __attribute__((ext_vector_type(4)))  int i32x4;

#if __has_builtin(__builtin_amdgcn_tensor_load_to_lds) && \
    __has_builtin(__builtin_amdgcn_s_wait_tensorcnt)
#define USE_TDM 1
#else
#define USE_TDM 0
#endif

static __device__ __forceinline__ unsigned short f32_to_bf16(float f) {
  union { float f; unsigned int u; } c; c.f = f;
  unsigned int u = c.u;
  u += 0x7fffu + ((u >> 16) & 1u);              // round-to-nearest-even
  return (unsigned short)(u >> 16);
}

// ---- W fp32 -> bf16 (once per call; 512x1024, row-major [h][f]) -----------
__global__ void wconv_kernel(const float4* __restrict__ W4,
                             unsigned short* __restrict__ Wbf) {
  unsigned t = blockIdx.x * blockDim.x + threadIdx.x;   // one float4 each
  float4 v = W4[t];
  ushort4 o;
  o.x = f32_to_bf16(v.x); o.y = f32_to_bf16(v.y);
  o.z = f32_to_bf16(v.z); o.w = f32_to_bf16(v.w);
  *(ushort4*)(Wbf + (size_t)t * 4) = o;
}

// ---- leaf gather + tanh + bf16 pack ---------------------------------------
// emb table (65MB) is L2-resident on a 192MB L2; gather reads hit cache.
__global__ void embed_tanh_kernel(const int* __restrict__ ids,
                                  const float4* __restrict__ emb4,
                                  unsigned short* __restrict__ h0) {
  unsigned t    = blockIdx.x * blockDim.x + threadIdx.x;  // one float4 each
  unsigned leaf = t >> 7;                                  // 128 float4 / row
  unsigned j    = t & 127;
  int id = ids[leaf];
  float4 v = emb4[(size_t)id * 128 + j];
  ushort4 o;
  o.x = f32_to_bf16(tanhf(v.x)); o.y = f32_to_bf16(tanhf(v.y));
  o.z = f32_to_bf16(tanhf(v.z)); o.w = f32_to_bf16(tanhf(v.w));
  *(ushort4*)(h0 + (size_t)leaf * HID + j * 4) = o;
}

#if USE_TDM
// Issue one TDM 2D-tile load: W[n0..n0+63][kc0..kc0+KC) -> LDS @ lds_byte_off.
// D# per CDNA5 ISA ch.8: group0 = {count, lds_addr, global_addr, type=2};
// group1 = {data_size=2B, pad 4 DWORDs after every 128 DWORDs (528B LDS rows),
//           tensor 1024x512, tile 256x64, dim0 stride 1024}; groups 2/3 = 0
// (2-D tile -> VADDR2/3 unused; extra operands passed as zeros).
static __device__ __forceinline__ void tdm_load_w_tile(
    const unsigned short* wbf, int n0, int kc0, unsigned lds_byte_off) {
  unsigned long long ga =
      (unsigned long long)(uintptr_t)(wbf + (size_t)n0 * KDIM + kc0);
  u32x4 g0;
  g0.x = 1u;                                   // count=1, user mode, no gather
  g0.y = lds_byte_off;                         // D#[63:32]  lds_addr (bytes)
  g0.z = (unsigned)(ga & 0xffffffffu);         // D#[95:64]  global_addr[31:0]
  g0.w = (unsigned)((ga >> 32) & 0x1ffffffu)   // D#[120:96] global_addr[56:32]
         | (2u << 30);                         // D#[127:126] type = 2 (image)
  i32x8 g1;
  g1[0] = (int)((1u << 16)       // data_size = 1 -> 2 bytes/element
              | (1u << 20)       // pad_enable
              | (6u << 22)       // pad_interval: 128 DWORDs (512B) per row
              | (3u << 25));     // pad_amount:   4 DWORDs (16B) -> 528B rows
  g1[1] = (int)(((unsigned)KDIM) << 16);  // tensor_dim0 = 1024 (elements)
  g1[2] = (int)(((unsigned)HID)  << 16);  // tensor_dim1 = 512
  g1[3] = (int)(((unsigned)KC)   << 16);  // tile_dim0   = 256
  g1[4] = 64;                             // tile_dim1   = 64 rows
  g1[5] = KDIM;                           // tensor_dim0_stride = 1024
  g1[6] = 0;
  g1[7] = 0;
  i32x4 z4 = {0, 0, 0, 0};
  i32x8 z8 = {0, 0, 0, 0, 0, 0, 0, 0};
  __builtin_amdgcn_tensor_load_to_lds(g0, g1, z4, z4, z8, 0);
}
#endif

// ---- one tree level: out[rows_out][512] = tanh(X[rows_out][1024] @ W^T + b)
// Block = 256 threads = 8 waves. Wave w -> M rows [blk*128 + w*16, +16),
// N cols [blockIdx.y*64, +64) as 4 accumulators of 16x16.
__global__ __launch_bounds__(256)
void tree_gemm_wmma_kernel(const unsigned short* __restrict__ hin_u,
                           const unsigned short* __restrict__ wbf_u,
                           const float* __restrict__ bias,
                           unsigned short* __restrict__ outb_u,
                           float* __restrict__ outf,
                           int rows_out) {
#if USE_TDM
  __shared__ unsigned short wlds_u[2 * WTILE_H];   // double-buffered W tile
#else
  __shared__ unsigned short wlds_u[WTILE_H];
#endif
  const bf16_t* hin  = (const bf16_t*)hin_u;
  bf16_t*       wlds = (bf16_t*)wlds_u;

  const int tid  = threadIdx.x;
  const int wave = tid >> 5;
  const int lane = tid & 31;
  const int nl   = lane & 15;      // N index in 16-subtile / A row in M-tile
  const int g    = lane >> 4;      // lane half-group

  const int m0 = blockIdx.x * 128 + wave * 16;
  const int n0 = blockIdx.y * 64;
  const bool active = (m0 < rows_out);        // wave-uniform (tail levels)

  // A fragment row for this lane (ISA 16-bit A 16x32 layout: M = lane&15)
  const bf16_t* arow = hin + (size_t)(m0 + nl) * KDIM;

  v8f acc[4] = {};   // f32 16x16 accumulators

#if USE_TDM
  if (wave == 0) tdm_load_w_tile(wbf_u, n0, 0, 0u);          // prime chunk 0
  int buf = 0;
#endif

  for (int ci = 0; ci < KDIM / KC; ++ci) {
    const int kc0 = ci * KC;
#if USE_TDM
    if (wave == 0) __builtin_amdgcn_s_wait_tensorcnt(0);     // chunk ci landed
    __syncthreads();                                         // visible to all
    // TDM wrote wlds_u behind the optimizer's back (descriptor carries the
    // LDS address as integers). Escape the pointer + full memory clobber so
    // the B-fragment ds_loads below are neither folded to undef nor hoisted.
    asm volatile("" : : "r"(wlds) : "memory");
    if (wave == 0 && kc0 + KC < KDIM)                        // DMA next chunk
      tdm_load_w_tile(wbf_u, n0, kc0 + KC,
                      (unsigned)((buf ^ 1) * WTILE_H * 2));
    const bf16_t* wtile = wlds + (size_t)buf * WTILE_H;
    buf ^= 1;
#else
    __syncthreads();
    #pragma unroll
    for (int u = tid; u < 64 * (KC / 8); u += 256) {         // fallback stage
      int nr   = u >> 5;
      int koff = (u & 31) * 8;
      *(v8bf*)(wlds + (size_t)nr * LDS_STRIDE + koff) =
          *(const v8bf*)((const bf16_t*)wbf_u +
                         (size_t)(n0 + nr) * KDIM + kc0 + koff);
    }
    __syncthreads();
    const bf16_t* wtile = wlds;
#endif

    if (active) {
      if (kc0 + KC < KDIM)                       // gfx1250 global_prefetch_b8
        __builtin_prefetch(arow + kc0 + KC + 8 * g, 0, 3);

      #pragma unroll
      for (int kk = 0; kk < KC; kk += 32) {
        // A frag: lane holds K = kc0+kk + {8g..8g+7, 16+8g..16+8g+7}
        const bf16_t* ap = arow + kc0 + kk + 8 * g;
        v8bf alo = *(const v8bf*)ap;
        v8bf ahi = *(const v8bf*)(ap + 16);
        v16bf afrag = __builtin_shufflevector(alo, ahi,
            0,1,2,3,4,5,6,7,8,9,10,11,12,13,14,15);
        #pragma unroll
        for (int j = 0; j < 4; ++j) {
          // B frag: lane holds K = {8g..8g+7, 16+8g..16+8g+7} for column 16*j+nl
          const bf16_t* bp = wtile + (size_t)(16 * j + nl) * LDS_STRIDE
                                   + kk + 8 * g;
          v8bf blo = *(const v8bf*)bp;
          v8bf bhi = *(const v8bf*)(bp + 16);
          v16bf bfrag = __builtin_shufflevector(blo, bhi,
              0,1,2,3,4,5,6,7,8,9,10,11,12,13,14,15);
          acc[j] = __builtin_amdgcn_wmma_f32_16x16x32_bf16(
              false, afrag, false, bfrag, (short)0, acc[j], false, false);
        }
      }
    }
  }

  if (!active) return;

  // epilogue: bias + tanh; D layout: VGPR r, lanes 0-15 -> M=r, 16-31 -> M=8+r
  #pragma unroll
  for (int j = 0; j < 4; ++j) {
    const int col = n0 + 16 * j + nl;
    const float bv = bias[col];
    #pragma unroll
    for (int r = 0; r < 8; ++r) {
      const int row = m0 + 8 * g + r;
      const float y = tanhf(acc[j][r] + bv);
      if (outf) outf[(size_t)row * HID + col] = y;                 // root level
      else      outb_u[(size_t)row * HID + col] = f32_to_bf16(y);  // bf16 level
    }
  }
}

// ---------------------------------------------------------------------------
extern "C" void kernel_launch(void* const* d_in, const int* in_sizes, int n_in,
                              void* d_out, int out_size, void* d_ws, size_t ws_size,
                              hipStream_t stream) {
  const int*   ids  = (const int*)d_in[0];     // [64,2048] int32
  const float* emb  = (const float*)d_in[1];   // [32000,512] f32
  const float* W    = (const float*)d_in[2];   // [512,1024] f32
  const float* bias = (const float*)d_in[3];   // [512] f32
  float* out = (float*)d_out;                  // [64,1,512] f32

  // ws layout: [Wbf 1MB][ping 128MB][pong 64MB]  (~193MB)
  unsigned char* ws = (unsigned char*)d_ws;
  unsigned short* Wbf  = (unsigned short*)ws;
  unsigned short* ping = (unsigned short*)(ws + ((size_t)1 << 20));
  unsigned short* pong = (unsigned short*)(ws + ((size_t)1 << 20)
                                              + (size_t)NLEAF * HID * 2);

  wconv_kernel<<<(HID * KDIM / 4) / 256, 256, 0, stream>>>(
      (const float4*)W, Wbf);
  embed_tanh_kernel<<<((size_t)NLEAF * (HID / 4)) / 256, 256, 0, stream>>>(
      ids, (const float4*)emb, ping);

  unsigned short* cur = ping;
  unsigned short* nxt = pong;
  int rows = NLEAF;
  for (int lvl = 0; lvl < 11; ++lvl) {
    const int rows_out = rows >> 1;
    const bool last = (rows_out == B_TREES);      // 64 roots -> f32 d_out
    dim3 grid((rows_out + 127) / 128, HID / 64, 1);
    tree_gemm_wmma_kernel<<<grid, 256, 0, stream>>>(
        cur, Wbf, bias,
        last ? (unsigned short*)nullptr : nxt,
        last ? out : (float*)nullptr,
        rows_out);
    unsigned short* t = cur; cur = nxt; nxt = t;
    rows = rows_out;
  }
}
